// LectinOracle_8022998909113
// MI455X (gfx1250) — matrix operand order repaired and checked by reading, not compiled, not run on hardware
//
#include <hip/hip_runtime.h>
#include <hip/hip_bf16.h>

typedef __attribute__((ext_vector_type(16))) _Float16 v16h;
typedef __attribute__((ext_vector_type(8)))  float    v8f;
typedef __attribute__((ext_vector_type(4)))  _Float16 h4;
typedef __attribute__((ext_vector_type(4)))  float    f4v;
typedef __attribute__((ext_vector_type(4)))  unsigned int u32x4;
typedef __attribute__((ext_vector_type(8)))  int      i32x8;
typedef __attribute__((ext_vector_type(4)))  int      i32x4;

#define HID    128
#define NPG    64
#define EPG    128
#define NB     1024
#define NNODES (NB*NPG)     // 65536
#define NEDGES (NB*EPG)     // 131072
#define PROTD  1280
#define SLOPE  0.01f
#define EPSBN  1e-5f
#define DMINC  (-11.355f)
#define DMAXC  (23.892f)

// ---------------- WMMA fragment loaders (CDNA5 16-bit layouts, ISA 7.12.2) ----
// A 16x32 (MxK): lanes 0-15 hold M=lane, lanes 16-31 same M with K offset +8;
// VGPR v packs K = 16*(v>>2) + 2*(v&3) (+8 for upper half-lanes).
__device__ __forceinline__ v16h frag_a(const _Float16* base, int row, int ld,
                                       int kk, int lane) {
  const int h = (lane >> 4) & 1;
  const _Float16* p = base + (long)row * ld + kk + 8 * h;
  v16h f;
#pragma unroll
  for (int v = 0; v < 8; ++v) {
    const int k = 16 * (v >> 2) + 2 * (v & 3);
    f[2 * v]     = p[k];
    f[2 * v + 1] = p[k + 1];
  }
  return f;
}

// B 32x16 (KxN): lane holds column N=lane&15; same K order as frag_a:
// VGPR v packs K = 16*(v>>2) + 2*(v&3) (+8 for upper half-lanes).
// We feed B = W^T with W stored row-major [N][K], so lane reads row n of W.
__device__ __forceinline__ v16h frag_b(const _Float16* base, int col, int ld,
                                       int kk, int lane) {
  const int h = (lane >> 4) & 1;
  const _Float16* p = base + (long)col * ld + kk + 8 * h;
  v16h f;
#pragma unroll
  for (int v = 0; v < 8; ++v) {
    const int k = 16 * (v >> 2) + 2 * (v & 3);
    f[2 * v]     = p[k];
    f[2 * v + 1] = p[k + 1];
  }
  return f;
}

// ---------------- small utility kernels ---------------------------------------
// vectorized f32 -> f16 convert: n4 = n/4 elements of float4 -> h4 (8B stores)
__global__ void cvt_f16_v4(const float* __restrict__ s, _Float16* __restrict__ d,
                           int n4) {
  int i = blockIdx.x * 256 + threadIdx.x;
  if (i >= n4) return;
  f4v v = ((const f4v*)s)[i];
  h4 o;
  o[0] = (_Float16)v[0]; o[1] = (_Float16)v[1];
  o[2] = (_Float16)v[2]; o[3] = (_Float16)v[3];
  ((h4*)d)[i] = o;
}

// copy [rows][scols] f32 -> [rows][dcols] f16, zero-padding cols >= scols
__global__ void cvt_pad(const float* __restrict__ s, _Float16* __restrict__ d,
                        int rows, int scols, int dcols) {
  int i = blockIdx.x * 256 + threadIdx.x;
  if (i >= rows * dcols) return;
  int r = i / dcols, c = i - r * dcols;
  d[i] = (c < scols) ? (_Float16)s[r * scols + c] : (_Float16)0.f;
}

// zero pad columns [c0, ld) of an f16 [rows][ld] matrix
__global__ void zero_pad_h(_Float16* __restrict__ d, int rows, int ld, int c0) {
  int w = ld - c0;
  int i = blockIdx.x * 256 + threadIdx.x;
  if (i >= rows * w) return;
  int r = i / w, c = c0 + (i - r * w);
  d[(long)r * ld + c] = (_Float16)0.f;
}

// x[n][f] = emb[nodes[n]][f], vectorized 4-wide (float4 load, 8B f16 store)
__global__ void embed_gather(const float* __restrict__ emb,
                             const int* __restrict__ nodes,
                             _Float16* __restrict__ x) {
  int i = blockIdx.x * 256 + threadIdx.x;        // over NNODES*HID/4
  if (i >= NNODES * (HID / 4)) return;
  int n = i >> 5, f4 = i & (HID / 4 - 1);
  f4v v = ((const f4v*)emb)[(long)nodes[n] * (HID / 4) + f4];
  h4 o;
  o[0] = (_Float16)v[0]; o[1] = (_Float16)v[1];
  o[2] = (_Float16)v[2]; o[3] = (_Float16)v[3];
  ((h4*)x)[i] = o;
}

// ---------------- generic WMMA GEMM: out = leaky(A @ W^T + bias) --------------
// A: f16 [M][lda], W: f16 [N][ldw] (row-major, ldw >= K), out f32 [M][ldo]
__global__ __launch_bounds__(32)
void wmma_gemm_bias_leaky(const _Float16* __restrict__ A, int lda,
                          const _Float16* __restrict__ W, int ldw,
                          const float* __restrict__ bias,
                          float* __restrict__ out, int ldo, int K) {
  const int lane = threadIdx.x;
  const int m0 = blockIdx.x * 16;
  const int n0 = blockIdx.y * 16;
  const int lm = lane & 15, h = lane >> 4;
  v8f acc = {};
  for (int kk = 0; kk < K; kk += 32) {
    v16h a = frag_a(A, m0 + lm, lda, kk, lane);
    v16h b = frag_b(W, n0 + lm, ldw, kk, lane);
    acc = __builtin_amdgcn_wmma_f32_16x16x32_f16(false, a, false, b,
                                                 (short)0, acc, false, false);
  }
  const float bn = bias[n0 + lm];
#pragma unroll
  for (int r = 0; r < 8; ++r) {
    float v = acc[r] + bn;
    v = (v >= 0.f) ? v : SLOPE * v;
    out[(long)(m0 + r + 8 * h) * ldo + n0 + lm] = v;
  }
}

// ---------------- BatchNorm1d (training-mode batch stats, biased var) ---------
__global__ __launch_bounds__(256)
void bn_col(const float* __restrict__ X, int M, int ldx,
            const float* __restrict__ g, const float* __restrict__ b,
            float* __restrict__ outf, int ldo,
            _Float16* __restrict__ outh, int ldh) {
  const int c = blockIdx.x, t = threadIdx.x;
  __shared__ float s1[256], s2[256];
  float sum = 0.f, sq = 0.f;
  for (int r = t; r < M; r += 256) {
    float v = X[(long)r * ldx + c];
    sum += v; sq += v * v;
  }
  s1[t] = sum; s2[t] = sq; __syncthreads();
  for (int o = 128; o > 0; o >>= 1) {
    if (t < o) { s1[t] += s1[t + o]; s2[t] += s2[t + o]; }
    __syncthreads();
  }
  const float mean = s1[0] / M;
  const float var  = s2[0] / M - mean * mean;
  const float sc = g[c] * rsqrtf(var + EPSBN);
  const float bb = b[c] - mean * sc;
  for (int r = t; r < M; r += 256) {
    float v = X[(long)r * ldx + c] * sc + bb;
    if (outf) outf[(long)r * ldo + c] = v;
    if (outh) outh[(long)r * ldh + c] = (_Float16)v;
  }
}

// ---------------- fused GraphConv block: one workgroup per 64-node graph ------
// y = leaky(segsum(x[src]) @ Wl^T + bl + x @ Wr^T); y *= tanh(y.pw/||pw||);
// emits x_out (f16) and [max|mean] pool for this graph.
// x tile (64x128 f16 = 16KB) is DMA'd into LDS by the Tensor Data Mover.
__global__ __launch_bounds__(256)
void gc_block(const _Float16* __restrict__ xin,
              const int* __restrict__ ei,                 // [2][NEDGES]
              const _Float16* __restrict__ Wl, const float* __restrict__ bl,
              const _Float16* __restrict__ Wr, const float* __restrict__ pw,
              _Float16* __restrict__ xout, float* __restrict__ pool) {
  __shared__ _Float16 xh[NPG * HID];      // 16 KB (TDM destination)
  __shared__ _Float16 aggh[NPG * HID];    // 16 KB (also edge staging / scores)
  __shared__ float    fbuf[NPG * HID];    // 32 KB (agg f32, then y f32)
  const int g = blockIdx.x;
  const int t = threadIdx.x;
  const int wave = t >> 5, lane = t & 31;

  // stage local edge endpoints in aggh space (not yet needed as f16)
  int* es = (int*)aggh;                   // es[0..127]=src, es[128..255]=dst
  {
    const int which = t >> 7, j = t & (EPG - 1);   // 256 threads = 2*EPG
    es[t] = ei[(long)which * NEDGES + g * EPG + j] - g * NPG;
  }
  for (int i = t; i < NPG * HID; i += 256) fbuf[i] = 0.f;

#if __has_builtin(__builtin_amdgcn_tensor_load_to_lds) && \
    __has_builtin(__builtin_amdgcn_s_wait_tensorcnt)
  // ---- TDM: one wave issues the 2D tensor DMA global->LDS (ISA ch.8 D#) ----
  if (wave == 0) {
    const unsigned long long ga =
        (unsigned long long)(uintptr_t)(xin + (size_t)g * NPG * HID);
    const unsigned int la = (unsigned int)(uintptr_t)xh;  // LDS byte offset
    u32x4 g0;
    g0[0] = 1u;                                    // count=1, user descriptor
    g0[1] = la;                                    // lds_addr
    g0[2] = (unsigned int)(ga & 0xffffffffu);      // global_addr[31:0]
    g0[3] = (unsigned int)((ga >> 32) & 0x1ffffffu) | (2u << 30);  // + type=2
    i32x8 g1;
    g1[0] = (1 << 16);          // data_size = 2 bytes
    g1[1] = (HID << 16);        // tensor_dim0 = 128   (bits 79:48 low half)
    g1[2] = (NPG << 16);        // tensor_dim1 = 64    (bits 111:80 low half)
    g1[3] = (HID << 16);        // tile_dim0 = 128     (bits 127:112)
    g1[4] = NPG;                // tile_dim1 = 64, tile_dim2 = 0
    g1[5] = HID;                // tensor_dim0_stride = 128 (elements)
    g1[6] = 0; g1[7] = 0;       // tensor_dim1_stride = 0 (2D tile)
    i32x4 z4 = {0, 0, 0, 0};    // groups 2/3 unused (<=3D)
    i32x8 z8 = {0, 0, 0, 0, 0, 0, 0, 0};
    __builtin_amdgcn_tensor_load_to_lds(g0, g1, z4, z4, z8, 0);
    __builtin_amdgcn_s_wait_tensorcnt(0);
  }
#else
  {
    const _Float16* xg = xin + (long)g * NPG * HID;
    for (int i = t; i < NPG * HID; i += 256) xh[i] = xg[i];
  }
#endif
  __syncthreads();

  // edge scatter-add entirely in LDS (ds_add_f32)
  for (int i = t; i < EPG * HID; i += 256) {
    const int j = i >> 7, f = i & (HID - 1);
    atomicAdd(&fbuf[es[EPG + j] * HID + f], (float)xh[es[j] * HID + f]);
  }
  __syncthreads();
  for (int i = t; i < NPG * HID; i += 256) aggh[i] = (_Float16)fbuf[i];
  __syncthreads();

  // 8 waves: wave = (mt in 0..3, ng in 0..1); each does 4 16x16 N-tiles.
  const int lm = lane & 15, h = lane >> 4;
  const int mt = wave & 3, ng = wave >> 2;
#pragma unroll
  for (int nt = 0; nt < 4; ++nt) {
    const int n0 = (ng * 4 + nt) * 16;
    v8f acc = {};
#pragma unroll
    for (int ks = 0; ks < 4; ++ks) {
      const int kk = ks * 32;
      v16h aA = frag_a(aggh, mt * 16 + lm, HID, kk, lane);
      v16h bL = frag_b(Wl, n0 + lm, HID, kk, lane);
      acc = __builtin_amdgcn_wmma_f32_16x16x32_f16(false, aA, false, bL,
                                                   (short)0, acc, false, false);
      v16h aX = frag_a(xh, mt * 16 + lm, HID, kk, lane);
      v16h bR = frag_b(Wr, n0 + lm, HID, kk, lane);
      acc = __builtin_amdgcn_wmma_f32_16x16x32_f16(false, aX, false, bR,
                                                   (short)0, acc, false, false);
    }
    const float bias = bl[n0 + lm];
#pragma unroll
    for (int r = 0; r < 8; ++r) {
      float v = acc[r] + bias;
      v = (v >= 0.f) ? v : SLOPE * v;
      fbuf[(mt * 16 + r + 8 * h) * HID + n0 + lm] = v;
    }
  }
  __syncthreads();

  // TopK(ratio=1) gating: score = tanh((y . pw) / ||pw||)
  float* scorep = (float*)aggh;           // aggh free after WMMA
  if (t < NPG) {
    float n2 = 0.f, dot = 0.f;
    for (int f = 0; f < HID; ++f) {
      const float w = pw[f];
      n2 += w * w;
      dot += fbuf[t * HID + f] * w;
    }
    scorep[t] = tanhf(dot * rsqrtf(n2));
  }
  __syncthreads();

  // gate + vectorized f16 writeback (8B stores)
  h4* xo4 = (h4*)(xout + (long)g * NPG * HID);
  for (int i4 = t; i4 < NPG * HID / 4; i4 += 256) {
    const int base = i4 * 4;
    const float sc = scorep[base >> 7];
    h4 o;
#pragma unroll
    for (int q = 0; q < 4; ++q) {
      const float v = fbuf[base + q] * sc;
      fbuf[base + q] = v;
      o[q] = (_Float16)v;
    }
    xo4[i4] = o;
  }
  __syncthreads();

  // fused [global_max | global_mean] pooling for this graph
  if (t < HID) {
    float mx = -3.4e38f, sm = 0.f;
    for (int r = 0; r < NPG; ++r) {
      const float v = fbuf[r * HID + t];
      mx = fmaxf(mx, v); sm += v;
    }
    pool[(long)g * 256 + t]        = mx;
    pool[(long)g * 256 + HID + t]  = sm * (1.f / NPG);
  }
}

// pooled sum of 3 blocks -> f16 columns [128..384) of h
__global__ void combine_pool(const float* __restrict__ p1,
                             const float* __restrict__ p2,
                             const float* __restrict__ p3,
                             _Float16* __restrict__ hh) {
  int i = blockIdx.x * 256 + threadIdx.x;
  if (i >= NB * 256) return;
  int r = i >> 8, c = i & 255;
  hh[(long)r * 384 + 128 + c] = (_Float16)(p1[i] + p2[i] + p3[i]);
}

// fc2 + sigmoid scale (8 dropout heads coincide at eval -> single head)
__global__ void head_kernel(const float* __restrict__ hbn,
                            const float* __restrict__ W,
                            const float* __restrict__ b,
                            float* __restrict__ out) {
  int r = blockIdx.x * 256 + threadIdx.x;
  if (r >= NB) return;
  float s = b[0];
  for (int f = 0; f < 64; ++f) s += hbn[r * 64 + f] * W[f];
  const float sig = 1.f / (1.f + expf(-s));
  out[r] = sig * (DMAXC - DMINC) + DMINC;
}

// ---------------- launcher ----------------------------------------------------
extern "C" void kernel_launch(void* const* d_in, const int* in_sizes, int n_in,
                              void* d_out, int out_size, void* d_ws, size_t ws_size,
                              hipStream_t stream) {
  const float* prot = (const float*)d_in[0];
  const int*   nodes = (const int*)d_in[1];
  const int*   ei    = (const int*)d_in[2];
  // d_in[3] = batch: implied by graph structure (64 nodes/graph), unused
  const float* emb  = (const float*)d_in[4];
  const float* Wp1  = (const float*)d_in[5];  const float* bp1  = (const float*)d_in[6];
  const float* gp1  = (const float*)d_in[7];  const float* btp1 = (const float*)d_in[8];
  const float* Wp2  = (const float*)d_in[9];  const float* bp2  = (const float*)d_in[10];
  const float* gp2  = (const float*)d_in[11]; const float* btp2 = (const float*)d_in[12];
  const float* Wlf[3] = {(const float*)d_in[13], (const float*)d_in[17], (const float*)d_in[21]};
  const float* blf[3] = {(const float*)d_in[14], (const float*)d_in[18], (const float*)d_in[22]};
  const float* Wrf[3] = {(const float*)d_in[15], (const float*)d_in[19], (const float*)d_in[23]};
  const float* pwf[3] = {(const float*)d_in[16], (const float*)d_in[20], (const float*)d_in[24]};
  const float* Wfc1 = (const float*)d_in[25]; const float* bfc1 = (const float*)d_in[26];
  const float* g1   = (const float*)d_in[27]; const float* bt1  = (const float*)d_in[28];
  const float* Wfc2 = (const float*)d_in[29]; const float* bfc2 = (const float*)d_in[30];
  float* out = (float*)d_out;

  // ---- workspace carve-up (~45 MB) ----
  char* ws = (char*)d_ws;
  size_t off = 0;
  auto alloc = [&](size_t bytes) -> void* {
    void* p = ws + off;
    off += (bytes + 255) & ~(size_t)255;
    return p;
  };
  _Float16* prot_h = (_Float16*)alloc((size_t)NB * PROTD * 2);
  _Float16* Wp1_h  = (_Float16*)alloc((size_t)400 * PROTD * 2);
  float*    y1     = (float*)   alloc((size_t)NB * 400 * 4);
  _Float16* ep1_h  = (_Float16*)alloc((size_t)NB * 416 * 2);   // K padded 400->416
  _Float16* Wp2_h  = (_Float16*)alloc((size_t)128 * 416 * 2);
  float*    y2     = (float*)   alloc((size_t)NB * 128 * 4);
  _Float16* h_h    = (_Float16*)alloc((size_t)NB * 384 * 2);   // [ep | pools]
  _Float16* Wl_h[3], *Wr_h[3];
  for (int i = 0; i < 3; ++i) {
    Wl_h[i] = (_Float16*)alloc((size_t)HID * HID * 2);
    Wr_h[i] = (_Float16*)alloc((size_t)HID * HID * 2);
  }
  _Float16* Wfc1_h = (_Float16*)alloc((size_t)64 * 384 * 2);
  _Float16* x_a    = (_Float16*)alloc((size_t)NNODES * HID * 2);
  _Float16* x_b    = (_Float16*)alloc((size_t)NNODES * HID * 2);
  float*    poolb[3];
  for (int i = 0; i < 3; ++i) poolb[i] = (float*)alloc((size_t)NB * 256 * 4);
  float*    fc1    = (float*)alloc((size_t)NB * 64 * 4);
  (void)ws_size; (void)in_sizes; (void)n_in; (void)out_size;

  auto g1d = [](int n) { return dim3((n + 255) / 256); };

  // ---- weight / input conversions to f16 (vectorized) ----
  cvt_f16_v4<<<g1d(NB * PROTD / 4), 256, 0, stream>>>(prot, prot_h, NB * PROTD / 4);
  cvt_f16_v4<<<g1d(400 * PROTD / 4), 256, 0, stream>>>(Wp1, Wp1_h, 400 * PROTD / 4);
  cvt_pad<<<g1d(128 * 416), 256, 0, stream>>>(Wp2, Wp2_h, 128, 400, 416);
  zero_pad_h<<<g1d(NB * 16), 256, 0, stream>>>(ep1_h, NB, 416, 400);
  for (int i = 0; i < 3; ++i) {
    cvt_f16_v4<<<g1d(HID * HID / 4), 256, 0, stream>>>(Wlf[i], Wl_h[i], HID * HID / 4);
    cvt_f16_v4<<<g1d(HID * HID / 4), 256, 0, stream>>>(Wrf[i], Wr_h[i], HID * HID / 4);
  }
  cvt_f16_v4<<<g1d(64 * 384 / 4), 256, 0, stream>>>(Wfc1, Wfc1_h, 64 * 384 / 4);

  // ---- protein encoder ----
  wmma_gemm_bias_leaky<<<dim3(NB / 16, 400 / 16), 32, 0, stream>>>(
      prot_h, PROTD, Wp1_h, PROTD, bp1, y1, 400, PROTD);
  bn_col<<<400, 256, 0, stream>>>(y1, NB, 400, gp1, btp1,
                                  (float*)nullptr, 0, ep1_h, 416);
  wmma_gemm_bias_leaky<<<dim3(NB / 16, 128 / 16), 32, 0, stream>>>(
      ep1_h, 416, Wp2_h, 416, bp2, y2, 128, 416);
  bn_col<<<128, 256, 0, stream>>>(y2, NB, 128, gp2, btp2,
                                  (float*)nullptr, 0, h_h, 384);  // h[:, 0:128]

  // ---- graph branch ----
  embed_gather<<<g1d(NNODES * HID / 4), 256, 0, stream>>>(emb, nodes, x_a);
  gc_block<<<NB, 256, 0, stream>>>(x_a, ei, Wl_h[0], blf[0], Wr_h[0], pwf[0],
                                   x_b, poolb[0]);
  gc_block<<<NB, 256, 0, stream>>>(x_b, ei, Wl_h[1], blf[1], Wr_h[1], pwf[1],
                                   x_a, poolb[1]);
  gc_block<<<NB, 256, 0, stream>>>(x_a, ei, Wl_h[2], blf[2], Wr_h[2], pwf[2],
                                   x_b, poolb[2]);
  combine_pool<<<g1d(NB * 256), 256, 0, stream>>>(poolb[0], poolb[1], poolb[2], h_h);

  // ---- head ----
  wmma_gemm_bias_leaky<<<dim3(NB / 16, 64 / 16), 32, 0, stream>>>(
      h_h, 384, Wfc1_h, 384, bfc1, fc1, 64, 384);
  bn_col<<<64, 256, 0, stream>>>(fc1, NB, 64, g1, bt1, fc1, 64,
                                 (_Float16*)nullptr, 0);
  head_kernel<<<g1d(NB), 256, 0, stream>>>(fc1, Wfc2, bfc2, out);
}
